// CausalTransformerLayer_16896401342947
// MI455X (gfx1250) — matrix operand order repaired and checked by reading, not compiled, not run on hardware
//
#include <hip/hip_runtime.h>
#include <hip/hip_bf16.h>

// ---------------- problem constants ----------------
constexpr int kB  = 4;
constexpr int kT  = 2048;
constexpr int kD  = 1024;
constexpr int kF  = 4096;
constexpr int kH  = 16;
constexpr int kHD = 64;
constexpr int kM  = kB * kT;          // 8192 rows of activations

typedef __attribute__((ext_vector_type(16))) __bf16 v16bf;
typedef __attribute__((ext_vector_type(8)))  __bf16 v8bf;
typedef __attribute__((ext_vector_type(8)))  float  v8f;

union bf16x16 { v16bf v; v8bf h[2]; };

__device__ __forceinline__ v8f wmma_bf16(v16bf a, v16bf b, v8f c) {
  // v_wmma_f32_16x16x32_bf16  (D = A*B + C, f32 accum)
  return __builtin_amdgcn_wmma_f32_16x16x32_bf16(false, a, false, b, (short)0, c,
                                                 false, false);
}

__device__ __forceinline__ v8f vzero8() { v8f z = {0,0,0,0,0,0,0,0}; return z; }

// low 32 bits of a generic LDS pointer == LDS byte offset (aperture bits live in [63:32])
__device__ __forceinline__ unsigned lds_off32(const void* p) {
  return (unsigned)(unsigned long long)p;
}

// async global -> LDS 16B copy (ASYNCcnt path, GV addressing); bypasses VGPR data regs
__device__ __forceinline__ void async_copy_b128(unsigned lds_dst, const void* gsrc) {
  asm volatile("global_load_async_to_lds_b128 %0, %1, off"
               :: "v"(lds_dst), "v"(gsrc) : "memory");
}
__device__ __forceinline__ void wait_asynccnt0() {
  asm volatile("s_wait_asynccnt 0x0" ::: "memory");
}

__device__ __forceinline__ float rowmax16(float v) {
  v = fmaxf(v, __shfl_xor(v, 1, 32));
  v = fmaxf(v, __shfl_xor(v, 2, 32));
  v = fmaxf(v, __shfl_xor(v, 4, 32));
  v = fmaxf(v, __shfl_xor(v, 8, 32));
  return v;
}
__device__ __forceinline__ float rowsum16(float v) {
  v += __shfl_xor(v, 1, 32);
  v += __shfl_xor(v, 2, 32);
  v += __shfl_xor(v, 4, 32);
  v += __shfl_xor(v, 8, 32);
  return v;
}

// ---------------- f32 -> bf16 weight conversion ----------------
__global__ __launch_bounds__(256) void cvt_bf16_kernel(const float* __restrict__ s,
                                                       __bf16* __restrict__ d, long n) {
  long i = (long)blockIdx.x * 256 + threadIdx.x;
  if (i < n) d[i] = (__bf16)s[i];
}

// ---------------- layernorm (one row of D=1024 per block) ----------------
__global__ __launch_bounds__(256) void ln_kernel(const float* __restrict__ x,
                                                 const float* __restrict__ sc,
                                                 const float* __restrict__ bi,
                                                 __bf16* __restrict__ out) {
  __shared__ float s1[256], s2[256];
  const int tid = threadIdx.x;
  const long row = blockIdx.x;
  const float* xr = x + row * kD;
  float v[4]; float sum = 0.f, sq = 0.f;
#pragma unroll
  for (int i = 0; i < 4; ++i) { v[i] = xr[tid + 256 * i]; sum += v[i]; sq += v[i] * v[i]; }
  s1[tid] = sum; s2[tid] = sq; __syncthreads();
  for (int st = 128; st > 0; st >>= 1) {
    if (tid < st) { s1[tid] += s1[tid + st]; s2[tid] += s2[tid + st]; }
    __syncthreads();
  }
  const float mu  = s1[0] * (1.f / kD);
  const float var = s2[0] * (1.f / kD) - mu * mu;
  const float inv = rsqrtf(var + 1e-6f);
  __bf16* orow = out + row * kD;
#pragma unroll
  for (int i = 0; i < 4; ++i) {
    const int idx = tid + 256 * i;
    orow[idx] = (__bf16)((v[i] - mu) * inv * sc[idx] + bi[idx]);
  }
}

// ---------------- shared 64x64x32 WMMA mainloop ----------------
// block = 128 threads = 4 waves; wave w computes rows [row0+16w, row0+16w+16) x 64 cols.
// A tile is staged by async DMA (ASYNCcnt); B tile staged+transposed through VGPRs,
// overlapping with the A DMA; one s_wait_asynccnt + barrier publishes the tile.
__device__ __forceinline__ void gemm_mainloop(const __bf16* __restrict__ A, long lda,
                                              const __bf16* __restrict__ B, long ldb,
                                              int K, long row0, long col0,
                                              __bf16* As, __bf16* Bs, v8f acc[4]) {
  const int tid    = threadIdx.x;
  const int lane   = tid & 31;
  const int wave   = tid >> 5;
  const int mloc   = lane & 15;
  const int half   = lane >> 4;
  const int khalfA = half * 16;   // A frag: contiguous 16-wide K halves
  const int khalfB = half * 16;   // B frag: contiguous 16-wide K halves

#pragma unroll
  for (int j = 0; j < 4; ++j) acc[j] = vzero8();

  for (int k0 = 0; k0 < K; k0 += 32) {
    // 1) kick off A tile 64x32 DMA: global -> LDS, no VGPR data staging
#pragma unroll
    for (int c = 0; c < 2; ++c) {
      const int chunk = tid * 2 + c;            // 0..255
      const int r  = chunk >> 2;                // 0..63
      const int cc = (chunk & 3) * 8;           // 0..24
      async_copy_b128(lds_off32(As + r * 32 + cc),
                      A + (row0 + r) * lda + k0 + cc);
    }
    // 2) stage B tile 32x64 under the A DMA, stored transposed Bs[n][k]
#pragma unroll
    for (int c = 0; c < 2; ++c) {
      const int chunk = tid * 2 + c;            // 0..255
      const int r  = chunk >> 3;                // k row 0..31
      const int cc = (chunk & 7) * 8;           // n offset 0..56
      v8bf vb = *(const v8bf*)(B + (long)(k0 + r) * ldb + col0 + cc);
#pragma unroll
      for (int e = 0; e < 8; ++e) Bs[(cc + e) * 32 + r] = vb[e];
    }
    if (k0 + 32 < K)  // L2-resident weights: cheap speculative prefetch of next K slab
      __builtin_prefetch(B + (long)(k0 + 32) * ldb + col0, 0, 1);
    wait_asynccnt0();
    __syncthreads();

    bf16x16 af;
    const __bf16* ap = As + (wave * 16 + mloc) * 32 + khalfA;
    af.h[0] = *(const v8bf*)(ap);
    af.h[1] = *(const v8bf*)(ap + 8);
#pragma unroll
    for (int j = 0; j < 4; ++j) {
      bf16x16 bf;
      const __bf16* bp = Bs + (j * 16 + mloc) * 32 + khalfB;
      bf.h[0] = *(const v8bf*)(bp);
      bf.h[1] = *(const v8bf*)(bp + 8);
      acc[j] = wmma_bf16(af.v, bf.v, acc[j]);
    }
    __syncthreads();
  }
}

// ---------------- QKV GEMM + RoPE epilogue ----------------
// C[m, n], n in [0,3D): sel = n/D (0:q 1:k 2:v). 64-wide n-blocks align with heads.
__global__ __launch_bounds__(128) void gemm_qkv_rope_kernel(
    const __bf16* __restrict__ h, const __bf16* __restrict__ wqkv,
    const float* __restrict__ cosb, const float* __restrict__ sinb,
    __bf16* __restrict__ qb, __bf16* __restrict__ kb, __bf16* __restrict__ vtb) {
  __shared__ __align__(16) __bf16 As[64 * 32];
  __shared__ __align__(16) __bf16 Bs[64 * 32];
  v8f acc[4];
  const long row0 = (long)blockIdx.y * 64;
  const long col0 = (long)blockIdx.x * 64;
  gemm_mainloop(h, kD, wqkv, 3 * kD, kD, row0, col0, As, Bs, acc);

  const int lane = threadIdx.x & 31;
  const int wave = threadIdx.x >> 5;
  const int mloc = lane & 15;
  const int half = lane >> 4;

  const int sel   = (int)(col0 >> 10);          // 0,1,2
  const int head  = (int)((col0 & 1023) >> 6);  // 0..15
#pragma unroll
  for (int r = 0; r < 8; ++r) {
    const long m = row0 + wave * 16 + r + 8 * half;
    const int  t = (int)(m & (kT - 1));
    const int  b = (int)(m >> 11);
    const long bh = (long)(b * kH + head);
    if (sel == 2) {
#pragma unroll
      for (int j = 0; j < 4; ++j) {
        const int hd = j * 16 + mloc;
        vtb[bh * kHD * kT + (long)hd * kT + t] = (__bf16)acc[j][r];
      }
    } else {
      const float c0 = cosb[(long)t * 32 + mloc];        // hd%32 in [0,16)
      const float s0 = sinb[(long)t * 32 + mloc];
      const float c1 = cosb[(long)t * 32 + 16 + mloc];   // hd%32 in [16,32)
      const float s1 = sinb[(long)t * 32 + 16 + mloc];
      __bf16* dst = (sel == 0) ? qb : kb;
      const float qs = (sel == 0) ? 0.125f : 1.0f;       // fold 1/sqrt(hd) into q
#pragma unroll
      for (int j = 0; j < 4; ++j) {
        const int hd = j * 16 + mloc;
        const float x  = acc[j][r];
        const float rot = (j < 2) ? -acc[j + 2][r] : acc[j - 2][r];
        const float c = (j & 1) ? c1 : c0;
        const float s = (j & 1) ? s1 : s0;
        dst[(bh * kT + t) * kHD + hd] = (__bf16)((x * c + rot * s) * qs);
      }
    }
  }
}

// ---------------- causal flash attention ----------------
// grid = (T/64, B*H), block = 128 (4 waves); wave owns 16 q rows of the 64-row block.
__global__ __launch_bounds__(128) void flash_attn_kernel(
    const __bf16* __restrict__ qg, const __bf16* __restrict__ kg,
    const __bf16* __restrict__ vtg, __bf16* __restrict__ attn) {
  __shared__ __align__(16) __bf16 Ps[4][16 * 64];   // per-wave P tile (C->A relayout)

  const int lane = threadIdx.x & 31;
  const int wave = threadIdx.x >> 5;
  const int mloc = lane & 15;
  const int half = lane >> 4;
  const int khalfA = half * 16;
  const int khalfB = half * 16;

  const int bh = blockIdx.y;
  const int b  = bh >> 4;
  const int head = bh & 15;
  const __bf16* qp  = qg  + (long)bh * kT * kHD;
  const __bf16* kp  = kg  + (long)bh * kT * kHD;
  const __bf16* vtp = vtg + (long)bh * kHD * kT;

  const int qblock = blockIdx.x;
  const int qbase  = qblock * 64;
  const int m0     = wave * 16;

  // Q fragments (A operand), resident for the whole loop
  bf16x16 qf[2];
#pragma unroll
  for (int s = 0; s < 2; ++s) {
    const __bf16* qpr = qp + (long)(qbase + m0 + mloc) * kHD + s * 32 + khalfA;
    qf[s].h[0] = *(const v8bf*)(qpr);
    qf[s].h[1] = *(const v8bf*)(qpr + 8);
  }

  const float NEG = -3.0e38f;
  float row_max[8], row_sum[8];
  v8f acc_o[4];
#pragma unroll
  for (int r = 0; r < 8; ++r) { row_max[r] = NEG; row_sum[r] = 0.f; }
#pragma unroll
  for (int j = 0; j < 4; ++j) acc_o[j] = vzero8();

  for (int jb = 0; jb <= qblock; ++jb) {
    const int kbase = jb * 64;
    // S = Q @ K^T (64 keys)
    v8f sacc[4];
#pragma unroll
    for (int j = 0; j < 4; ++j) sacc[j] = vzero8();
#pragma unroll
    for (int j = 0; j < 4; ++j) {
#pragma unroll
      for (int s = 0; s < 2; ++s) {
        bf16x16 kf;
        const __bf16* kpr = kp + (long)(kbase + j * 16 + mloc) * kHD + s * 32 + khalfB;
        kf.h[0] = *(const v8bf*)(kpr);
        kf.h[1] = *(const v8bf*)(kpr + 8);
        sacc[j] = wmma_bf16(qf[s].v, kf.v, sacc[j]);
      }
    }
    // causal mask on the diagonal block
    if (jb == qblock) {
#pragma unroll
      for (int j = 0; j < 4; ++j) {
        const int col = j * 16 + mloc;
#pragma unroll
        for (int r = 0; r < 8; ++r) {
          const int row = m0 + r + 8 * half;
          if (col > row) sacc[j][r] = NEG;
        }
      }
    }
    // online softmax (per-row state lives per r; rows span 16 lanes of a half)
#pragma unroll
    for (int r = 0; r < 8; ++r) {
      float mt = fmaxf(fmaxf(sacc[0][r], sacc[1][r]), fmaxf(sacc[2][r], sacc[3][r]));
      mt = rowmax16(mt);
      const float mnew = fmaxf(row_max[r], mt);
      const float scale = __expf(row_max[r] - mnew);
      row_sum[r] *= scale;
#pragma unroll
      for (int j = 0; j < 4; ++j) acc_o[j][r] *= scale;
      float ps = 0.f;
#pragma unroll
      for (int j = 0; j < 4; ++j) {
        const float p = __expf(sacc[j][r] - mnew);
        sacc[j][r] = p;
        ps += p;
      }
      row_sum[r] += rowsum16(ps);
      row_max[r] = mnew;
    }
    // C-layout -> A-layout relayout of P through LDS
#pragma unroll
    for (int j = 0; j < 4; ++j)
#pragma unroll
      for (int r = 0; r < 8; ++r)
        Ps[wave][(r + 8 * half) * 64 + j * 16 + mloc] = (__bf16)sacc[j][r];
    __syncthreads();
    // O += P @ V  (V pre-transposed: contiguous along keys)
#pragma unroll
    for (int s = 0; s < 2; ++s) {
      bf16x16 pf;
      const __bf16* pr = &Ps[wave][mloc * 64 + s * 32 + khalfA];
      pf.h[0] = *(const v8bf*)(pr);
      pf.h[1] = *(const v8bf*)(pr + 8);
#pragma unroll
      for (int j = 0; j < 4; ++j) {
        bf16x16 vf;
        const __bf16* vr = vtp + (long)(j * 16 + mloc) * kT + kbase + s * 32 + khalfB;
        vf.h[0] = *(const v8bf*)(vr);
        vf.h[1] = *(const v8bf*)(vr + 8);
        acc_o[j] = wmma_bf16(pf.v, vf.v, acc_o[j]);
      }
    }
    __syncthreads();
  }
  // normalize + store [b, t, head, hd] as bf16 for the Wo GEMM
#pragma unroll
  for (int r = 0; r < 8; ++r) {
    const int t = qbase + m0 + r + 8 * half;
    const float inv = 1.0f / row_sum[r];
#pragma unroll
    for (int j = 0; j < 4; ++j) {
      const int hd = j * 16 + mloc;
      attn[((long)(b * kT + t)) * kD + head * kHD + hd] = (__bf16)(acc_o[j][r] * inv);
    }
  }
}

// ---------------- Wo GEMM + residual * gamma_attn -> x2 (f32) ----------------
__global__ __launch_bounds__(128) void gemm_wo_kernel(
    const __bf16* __restrict__ a, const __bf16* __restrict__ wo,
    const float* __restrict__ xres, const float* __restrict__ gamma,
    float* __restrict__ x2) {
  __shared__ __align__(16) __bf16 As[64 * 32];
  __shared__ __align__(16) __bf16 Bs[64 * 32];
  v8f acc[4];
  const long row0 = (long)blockIdx.y * 64;
  const long col0 = (long)blockIdx.x * 64;
  gemm_mainloop(a, kD, wo, kD, kD, row0, col0, As, Bs, acc);

  const int lane = threadIdx.x & 31;
  const int wave = threadIdx.x >> 5;
#pragma unroll
  for (int r = 0; r < 8; ++r) {
    const long m = row0 + wave * 16 + r + 8 * (lane >> 4);
#pragma unroll
    for (int j = 0; j < 4; ++j) {
      const long n = col0 + j * 16 + (lane & 15);
      x2[m * kD + n] = xres[m * kD + n] + acc[j][r] * gamma[n];
    }
  }
}

// ---------------- gate & up GEMMs + GELU(g)*u -> m (bf16) ----------------
__global__ __launch_bounds__(128) void gemm_gateup_kernel(
    const __bf16* __restrict__ h2, const __bf16* __restrict__ wg,
    const __bf16* __restrict__ wu, __bf16* __restrict__ mout) {
  __shared__ __align__(16) __bf16 As[64 * 32];
  __shared__ __align__(16) __bf16 Bs[64 * 32];
  v8f accg[4], accu[4];
  const long row0 = (long)blockIdx.y * 64;
  const long col0 = (long)blockIdx.x * 64;
  gemm_mainloop(h2, kD, wg, kF, kD, row0, col0, As, Bs, accg);
  gemm_mainloop(h2, kD, wu, kF, kD, row0, col0, As, Bs, accu);

  const int lane = threadIdx.x & 31;
  const int wave = threadIdx.x >> 5;
#pragma unroll
  for (int r = 0; r < 8; ++r) {
    const long m = row0 + wave * 16 + r + 8 * (lane >> 4);
#pragma unroll
    for (int j = 0; j < 4; ++j) {
      const long n = col0 + j * 16 + (lane & 15);
      const float g = accg[j][r];
      const float u = accu[j][r];
      const float gel = 0.5f * g * (1.0f + tanhf(0.7978845608f * (g + 0.044715f * g * g * g)));
      mout[m * kF + n] = (__bf16)(gel * u);
    }
  }
}

// ---------------- down GEMM + residual * gamma_mlp -> out (f32) ----------------
__global__ __launch_bounds__(128) void gemm_down_kernel(
    const __bf16* __restrict__ mact, const __bf16* __restrict__ wd,
    const float* __restrict__ x2, const float* __restrict__ gamma,
    float* __restrict__ out) {
  __shared__ __align__(16) __bf16 As[64 * 32];
  __shared__ __align__(16) __bf16 Bs[64 * 32];
  v8f acc[4];
  const long row0 = (long)blockIdx.y * 64;
  const long col0 = (long)blockIdx.x * 64;
  gemm_mainloop(mact, kF, wd, kD, kF, row0, col0, As, Bs, acc);

  const int lane = threadIdx.x & 31;
  const int wave = threadIdx.x >> 5;
#pragma unroll
  for (int r = 0; r < 8; ++r) {
    const long m = row0 + wave * 16 + r + 8 * (lane >> 4);
#pragma unroll
    for (int j = 0; j < 4; ++j) {
      const long n = col0 + j * 16 + (lane & 15);
      out[m * kD + n] = x2[m * kD + n] + acc[j][r] * gamma[n];
    }
  }
}

// ---------------- host-side orchestration ----------------
extern "C" void kernel_launch(void* const* d_in, const int* in_sizes, int n_in,
                              void* d_out, int out_size, void* d_ws, size_t ws_size,
                              hipStream_t stream) {
  const float* x        = (const float*)d_in[0];
  const float* cosb     = (const float*)d_in[1];
  const float* sinb     = (const float*)d_in[2];
  const float* ln1_s    = (const float*)d_in[3];
  const float* ln1_b    = (const float*)d_in[4];
  const float* w_qkv    = (const float*)d_in[5];
  const float* w_o      = (const float*)d_in[6];
  const float* g_attn   = (const float*)d_in[7];
  const float* ln2_s    = (const float*)d_in[8];
  const float* ln2_b    = (const float*)d_in[9];
  const float* w_gate   = (const float*)d_in[10];
  const float* w_up     = (const float*)d_in[11];
  const float* w_down   = (const float*)d_in[12];
  const float* g_mlp    = (const float*)d_in[13];
  float* outp           = (float*)d_out;

  // workspace layout (bytes)
  char* p = (char*)d_ws;
  __bf16* wqkv_b = (__bf16*)p; p += (size_t)kD * 3 * kD * 2;       // 6.29 MB
  __bf16* wo_b   = (__bf16*)p; p += (size_t)kD * kD * 2;           // 2.10 MB
  __bf16* wg_b   = (__bf16*)p; p += (size_t)kD * kF * 2;           // 8.39 MB
  __bf16* wu_b   = (__bf16*)p; p += (size_t)kD * kF * 2;           // 8.39 MB
  __bf16* wd_b   = (__bf16*)p; p += (size_t)kF * kD * 2;           // 8.39 MB
  char* act0 = p;                                                   // 67.1 MB region
  const size_t ACT = (size_t)kM * kD * 2;                           // 16.78 MB each
  __bf16* h_bf   = (__bf16*)(act0);
  __bf16* q_bf   = (__bf16*)(act0 + ACT);
  __bf16* k_bf   = (__bf16*)(act0 + 2 * ACT);
  __bf16* vt_bf  = (__bf16*)(act0 + 3 * ACT);
  __bf16* m_bf   = (__bf16*)(act0);     // aliases h/q/k/vT (dead by then): 8192*4096*2 = 4*ACT
  p = act0 + 4 * ACT;
  __bf16* attn_bf = (__bf16*)p; p += ACT;
  float*  x2      = (float*)p;  p += (size_t)kM * kD * 4;          // 33.55 MB
  __bf16* h2_bf   = (__bf16*)p; p += ACT;

  // 1) weights f32 -> bf16
  {
    long n;
    n = (long)kD * 3 * kD; cvt_bf16_kernel<<<(n + 255) / 256, 256, 0, stream>>>(w_qkv,  wqkv_b, n);
    n = (long)kD * kD;     cvt_bf16_kernel<<<(n + 255) / 256, 256, 0, stream>>>(w_o,    wo_b,   n);
    n = (long)kD * kF;     cvt_bf16_kernel<<<(n + 255) / 256, 256, 0, stream>>>(w_gate, wg_b,   n);
    n = (long)kD * kF;     cvt_bf16_kernel<<<(n + 255) / 256, 256, 0, stream>>>(w_up,   wu_b,   n);
    n = (long)kF * kD;     cvt_bf16_kernel<<<(n + 255) / 256, 256, 0, stream>>>(w_down, wd_b,   n);
  }
  // 2) LN1
  ln_kernel<<<kM, 256, 0, stream>>>(x, ln1_s, ln1_b, h_bf);
  // 3) QKV + RoPE (N = 3072)
  gemm_qkv_rope_kernel<<<dim3(3 * kD / 64, kM / 64), 128, 0, stream>>>(
      h_bf, wqkv_b, cosb, sinb, q_bf, k_bf, vt_bf);
  // 4) causal flash attention
  flash_attn_kernel<<<dim3(kT / 64, kB * kH), 128, 0, stream>>>(q_bf, k_bf, vt_bf, attn_bf);
  // 5) Wo + residual
  gemm_wo_kernel<<<dim3(kD / 64, kM / 64), 128, 0, stream>>>(attn_bf, wo_b, x, g_attn, x2);
  // 6) LN2, gate/up + GELU
  ln_kernel<<<kM, 256, 0, stream>>>(x2, ln2_s, ln2_b, h2_bf);
  gemm_gateup_kernel<<<dim3(kF / 64, kM / 64), 128, 0, stream>>>(h2_bf, wg_b, wu_b, m_bf);
  // 7) down + residual -> out
  gemm_down_kernel<<<dim3(kD / 64, kM / 64), 128, 0, stream>>>(m_bf, wd_b, x2, g_mlp, outp);
}
